// GraphNet_68058051772526
// MI455X (gfx1250) — matrix operand order repaired and checked by reading, not compiled, not run on hardware
//
#include <hip/hip_runtime.h>
#include <hip/hip_bf16.h>
#include <stdint.h>

// ---------------------------------------------------------------------------
// CDNA5 (gfx1250) wave32 WMMA types
// ---------------------------------------------------------------------------
typedef __attribute__((ext_vector_type(16))) _Float16 v16h;
typedef __attribute__((ext_vector_type(8)))  float    v8f;
typedef __attribute__((ext_vector_type(4)))  unsigned int u32x4;
typedef __attribute__((ext_vector_type(8)))  int      i32x8;
typedef __attribute__((ext_vector_type(4)))  int      i32x4;

union HFrag { v16h v; _Float16 h[16]; };

// TDM availability (arity differs: clang-22/ROCm7.2 = 5 args, clang-23 = 6 args)
#if defined(__has_builtin)
#if __has_builtin(__builtin_amdgcn_tensor_load_to_lds) && \
    __has_builtin(__builtin_amdgcn_s_wait_tensorcnt)
#define ATHENA_TDM 1
#endif
#endif

#ifdef ATHENA_TDM
// Issue a 1-D TDM copy of `bytes` (multiple of 8) from global `gsrc` into LDS
// offset `ldsByteOff`. Descriptor packing per CDNA5 ISA.
__device__ inline void tdm_load_to_lds(const void* gsrc, unsigned ldsByteOff,
                                       unsigned bytes) {
    unsigned n8 = bytes >> 3;  // 8-byte elements (data_size=3)
    unsigned long long ga = (unsigned long long)(uintptr_t)gsrc;
    u32x4 g0;
    g0[0] = 1u;                                   // count=1, user mode
    g0[1] = ldsByteOff;                           // lds_addr
    g0[2] = (unsigned)(ga & 0xFFFFFFFFu);         // global_addr[31:0]
    g0[3] = (unsigned)((ga >> 32) & 0x01FFFFFFu)  // global_addr[56:32]
            | (2u << 30);                         // type=2 ("image")
    i32x8 g1;
    g1[0] = (int)(3u << 16);                      // data_size=3 (8B), mask=0
    g1[1] = (int)((n8 & 0xFFFFu) << 16);          // tensor_dim0[15:0] @ bits 63:48
    g1[2] = (int)((n8 >> 16) | (1u << 16));       // tensor_dim0[31:16] | tensor_dim1=1
    g1[3] = (int)((n8 & 0xFFFFu) << 16);          // tile_dim0 = n8 @ bits 127:112
    g1[4] = 0;                                    // tile_dim1=0, tile_dim2=0 (1-D)
    g1[5] = (int)n8;                              // tensor_dim0_stride[31:0]
    g1[6] = 0;
    g1[7] = 0;
    i32x4 z4 = {0, 0, 0, 0};
#if __clang_major__ >= 23
    i32x8 z8 = {0, 0, 0, 0, 0, 0, 0, 0};
    __builtin_amdgcn_tensor_load_to_lds(g0, g1, z4, z4, z8, 0);
#else
    __builtin_amdgcn_tensor_load_to_lds(g0, g1, z4, z4, 0);
#endif
}
#endif

// ---------------------------------------------------------------------------
// small utility kernels
// ---------------------------------------------------------------------------
__global__ void k_fill_f32(float* p, float v, int n) {
    int t = blockIdx.x * blockDim.x + threadIdx.x;
    if (t < n) p[t] = v;
}
__global__ void k_fill_u32(unsigned* p, unsigned v, int n) {
    int t = blockIdx.x * blockDim.x + threadIdx.x;
    if (t < n) p[t] = v;
}
__global__ void k_fill_i32(int* p, int v, int n) {
    int t = blockIdx.x * blockDim.x + threadIdx.x;
    if (t < n) p[t] = v;
}
__global__ void k_edges_init(const long long* __restrict__ ei, int* __restrict__ src,
                             int* __restrict__ dst, int* __restrict__ em, int E) {
    int t = blockIdx.x * blockDim.x + threadIdx.x;
    if (t < E) {
        src[t] = (int)ei[t];
        dst[t] = (int)ei[E + t];
        em[t]  = 1;
    }
}

// pre-transpose + convert weights: W[din,dout] f32 -> Wt[dout,din] f16 (once/launch)
__global__ void k_prep_wt(const float* __restrict__ W, _Float16* __restrict__ Wt,
                          int din, int dout) {
    int t = blockIdx.x * blockDim.x + threadIdx.x;
    if (t >= din * dout) return;
    int kk = t / dout, nn = t - kk * dout;
    Wt[nn * din + kk] = (_Float16)W[t];
}

// monotone float <-> uint encoding for atomic float max
__device__ inline unsigned f2ord(float f) {
    unsigned u = __float_as_uint(f);
    return (u & 0x80000000u) ? ~u : (u | 0x80000000u);
}
__device__ inline float ord2f(unsigned u) {
    return (u & 0x80000000u) ? __uint_as_float(u & 0x7FFFFFFFu) : __uint_as_float(~u);
}

// ---------------------------------------------------------------------------
// WMMA linear: Y[n,DOUT] = X[n,DIN] @ W[DIN,DOUT] + b   (optional ReLU)
// Wt: pre-transposed f16 weights [DOUT][DIN], staged to LDS via TDM (or copy).
// 256 threads = 8 waves; each wave computes a 16-row x DOUT tile.
// ---------------------------------------------------------------------------
template <int DIN, int DOUT, bool RELU>
__global__ __launch_bounds__(256) void k_linear_wmma(const float* __restrict__ X,
                                                     const _Float16* __restrict__ Wt,
                                                     const float* __restrict__ bias,
                                                     float* __restrict__ Y, int n) {
    __shared__ _Float16 sW[DOUT * DIN];  // first (only) LDS object -> offset 0
#ifdef ATHENA_TDM
    if (threadIdx.x < 32) {  // one wave issues the DMA (EXEC ignored by TDM)
        tdm_load_to_lds((const void*)Wt, 0u, (unsigned)(DOUT * DIN * 2));
        __builtin_amdgcn_s_wait_tensorcnt(0);
    }
#else
    for (int i = threadIdx.x; i < DOUT * DIN; i += 256) sW[i] = Wt[i];
#endif
    __syncthreads();

    const int tid  = threadIdx.x;
    const int wave = tid >> 5;
    const int lane = tid & 31;
    const int row0 = blockIdx.x * 128 + wave * 16;
    int m  = row0 + (lane & 15);
    int mc = (m < n) ? m : (n - 1);                 // clamp: keep EXEC all-ones
    const float* Xrow = X + (size_t)mc * DIN;
    const int kb8  = (lane & 16) ? 8 : 0;           // A: K layout per ISA 16-bit A 16x32
    const int kbB  = (lane & 16) ? 8 : 0;           // B: same K layout as A (kbB+0..7, kbB+16..23)
    const int nsub = lane & 15;

    v8f acc[DOUT / 16];
#pragma unroll
    for (int c = 0; c < DOUT / 16; c++) {
        v8f z;
#pragma unroll
        for (int e = 0; e < 8; e++) z[e] = 0.f;
        acc[c] = z;
    }

#pragma unroll
    for (int kt = 0; kt < DIN / 32; kt++) {
        HFrag a;
        const float* pa = Xrow + kt * 32 + kb8;
#pragma unroll
        for (int i = 0; i < 8; i++) a.h[i] = (_Float16)pa[i];          // K = kb8+0..7
#pragma unroll
        for (int i = 0; i < 8; i++) a.h[8 + i] = (_Float16)pa[16 + i]; // K = kb8+16..23
#pragma unroll
        for (int c = 0; c < DOUT / 16; c++) {
            HFrag b;
            const _Float16* pb = &sW[(size_t)(c * 16 + nsub) * DIN + kt * 32 + kbB];
#pragma unroll
            for (int i = 0; i < 8; i++) { b.h[i] = pb[i]; b.h[8 + i] = pb[16 + i]; }
            acc[c] = __builtin_amdgcn_wmma_f32_16x16x32_f16(
                false, a.v, false, b.v, (short)0, acc[c], false, false);
        }
    }

    // C/D layout: lane&15 = N, VGPR r -> M = r + 8*(lane>=16)
    const int rHi = (lane & 16) ? 8 : 0;
#pragma unroll
    for (int c = 0; c < DOUT / 16; c++) {
        int ncol = c * 16 + nsub;
        float bv = bias[ncol];
#pragma unroll
        for (int r = 0; r < 8; r++) {
            int mr = row0 + rHi + r;
            if (mr < n) {
                float vv = acc[c][r] + bv;
                if (RELU) vv = vv > 0.f ? vv : 0.f;
                Y[(size_t)mr * DOUT + ncol] = vv;
            }
        }
    }
}

// ---------------------------------------------------------------------------
// residual + layernorm: Y = LN(A + B) * g + b   (one wave per row of 128)
// ---------------------------------------------------------------------------
__global__ __launch_bounds__(256) void k_ln_res(const float* __restrict__ A,
                                                const float* __restrict__ Bm,
                                                const float* __restrict__ g,
                                                const float* __restrict__ b,
                                                float* __restrict__ Y, int n) {
    int row = blockIdx.x * 8 + (threadIdx.x >> 5);
    if (row >= n) return;
    int lane = threadIdx.x & 31;
    size_t base = (size_t)row * 128;
    float v0[4];
    float s = 0.f;
#pragma unroll
    for (int j = 0; j < 4; j++) {
        int c = j * 32 + lane;
        v0[j] = A[base + c] + Bm[base + c];
        s += v0[j];
    }
#pragma unroll
    for (int o = 16; o > 0; o >>= 1) s += __shfl_xor(s, o, 32);
    float mean = s * (1.f / 128.f);
    float q = 0.f;
#pragma unroll
    for (int j = 0; j < 4; j++) { float d = v0[j] - mean; q += d * d; }
#pragma unroll
    for (int o = 16; o > 0; o >>= 1) q += __shfl_xor(q, o, 32);
    float inv = rsqrtf(q * (1.f / 128.f) + 1e-5f);
#pragma unroll
    for (int j = 0; j < 4; j++) {
        int c = j * 32 + lane;
        Y[base + c] = (v0[j] - mean) * inv * g[c] + b[c];
    }
}

// ---------------------------------------------------------------------------
// attention: per (edge, head) alpha + segment max
// ---------------------------------------------------------------------------
__global__ void k_attn_alpha(const float* __restrict__ q, const float* __restrict__ k,
                             const int* __restrict__ src, const int* __restrict__ dst,
                             const int* __restrict__ em, float* __restrict__ alpha,
                             unsigned* __restrict__ amax, int E) {
    int t = blockIdx.x * blockDim.x + threadIdx.x;
    if (t >= E * 4) return;
    int e = t >> 2, h = t & 3;
    if (!em[e]) return;
    int s = src[e], d = dst[e];
    const float* qp = q + (size_t)d * 128 + h * 32;
    const float* kp = k + (size_t)s * 128 + h * 32;
    float a = 0.f;
#pragma unroll
    for (int i = 0; i < 32; i++) a += qp[i] * kp[i];
    a *= 0.1767766952966369f;  // 1/sqrt(32)
    alpha[t] = a;
    atomicMax(&amax[d * 4 + h], f2ord(a));
}

// ex = exp(alpha - amax_eff); den[dst] += ex  (alpha buffer rewritten in place)
__global__ void k_attn_exp(float* __restrict__ alpha, const unsigned* __restrict__ amax,
                           float* __restrict__ den, const int* __restrict__ dst,
                           const int* __restrict__ em, int E) {
    int t = blockIdx.x * blockDim.x + threadIdx.x;
    if (t >= E * 4) return;
    int e = t >> 2, h = t & 3;
    float ex = 0.f;
    if (em[e]) {
        float am = ord2f(amax[dst[e] * 4 + h]);
        if (!(am > -1e29f)) am = 0.f;
        ex = expf(alpha[t] - am);
        atomicAdd(&den[dst[e] * 4 + h], ex);
    }
    alpha[t] = ex;
}

// out[dst] += v[src] * (ex/den)   (one wave per edge; 4 heads x 32 ch)
__global__ __launch_bounds__(256) void k_attn_scatter(const float* __restrict__ ex,
                                                      const float* __restrict__ den,
                                                      const float* __restrict__ v,
                                                      const int* __restrict__ src,
                                                      const int* __restrict__ dst,
                                                      const int* __restrict__ em,
                                                      float* __restrict__ out, int E) {
    int e = blockIdx.x * 8 + (threadIdx.x >> 5);
    if (e >= E) return;
    if (!em[e]) return;
    int lane = threadIdx.x & 31;
    int s = src[e], d = dst[e];
#pragma unroll
    for (int h = 0; h < 4; h++) {
        float w = ex[e * 4 + h] / (den[d * 4 + h] + 1e-16f);
        int c = h * 32 + lane;
        atomicAdd(&out[(size_t)d * 128 + c], v[(size_t)s * 128 + c] * w);
    }
}

// GraphConv aggregation for SAGPool scoring: agg[dst] += x[src]
__global__ __launch_bounds__(256) void k_pool_scatter(const float* __restrict__ x,
                                                      const int* __restrict__ src,
                                                      const int* __restrict__ dst,
                                                      const int* __restrict__ em,
                                                      float* __restrict__ agg, int E) {
    int e = blockIdx.x * 8 + (threadIdx.x >> 5);
    if (e >= E) return;
    if (!em[e]) return;
    int lane = threadIdx.x & 31;
    int s = src[e], d = dst[e];
#pragma unroll
    for (int j = 0; j < 4; j++) {
        int c = j * 32 + lane;
        atomicAdd(&agg[(size_t)d * 128 + c], x[(size_t)s * 128 + c]);
    }
}

// score = agg . relw + relb + x . rootw + rootb   (wave per node)
__global__ void k_score(const float* __restrict__ agg, const float* __restrict__ x,
                        const float* __restrict__ relw, const float* __restrict__ relb,
                        const float* __restrict__ rootw, const float* __restrict__ rootb,
                        float* __restrict__ score, int n) {
    int node = blockIdx.x * 8 + (threadIdx.x >> 5);
    if (node >= n) return;
    int lane = threadIdx.x & 31;
    size_t base = (size_t)node * 128;
    float p = 0.f;
#pragma unroll
    for (int j = 0; j < 4; j++) {
        int c = j * 32 + lane;
        p += agg[base + c] * relw[c] + x[base + c] * rootw[c];
    }
#pragma unroll
    for (int o = 16; o > 0; o >>= 1) p += __shfl_xor(p, o, 32);
    if (lane == 0) score[node] = p + relb[0] + rootb[0];
}

// exact per-graph top-k: bitonic sort (desc score, asc idx) of <=2048 keys in LDS
__global__ __launch_bounds__(256) void k_topk(const float* __restrict__ score, int NP,
                                              int KSEL, int* __restrict__ perm) {
    __shared__ unsigned long long keys[2048];
    int g = blockIdx.x, tid = threadIdx.x;
    for (int i = tid; i < 2048; i += 256) {
        unsigned long long kv = 0ull;  // pad sinks to end (finite enc > 0)
        if (i < NP) {
            float sc = score[g * NP + i];
            kv = ((unsigned long long)f2ord(sc) << 32) | (unsigned)(~(unsigned)i);
        }
        keys[i] = kv;
    }
    __syncthreads();
    for (int ksz = 2; ksz <= 2048; ksz <<= 1) {
        for (int j = ksz >> 1; j > 0; j >>= 1) {
            for (int i = tid; i < 2048; i += 256) {
                int ixj = i ^ j;
                if (ixj > i) {
                    unsigned long long a = keys[i], bb = keys[ixj];
                    bool up = ((i & ksz) == 0);
                    bool sw = up ? (a < bb) : (a > bb);  // overall descending
                    if (sw) { keys[i] = bb; keys[ixj] = a; }
                }
            }
            __syncthreads();
        }
    }
    for (int p = tid; p < KSEL; p += 256) {
        unsigned lidx = ~(unsigned)(keys[p] & 0xFFFFFFFFull);
        perm[g * KSEL + p] = g * NP + (int)lidx;
    }
}

// xn[p] = x[perm[p]] * tanh(score[perm[p]])
__global__ void k_gather_tanh(const float* __restrict__ x, const float* __restrict__ score,
                              const int* __restrict__ perm, float* __restrict__ xn, int nNew) {
    int t = blockIdx.x * blockDim.x + threadIdx.x;
    if (t >= nNew * 128) return;
    int p = t >> 7, c = t & 127;
    int node = perm[p];
    xn[t] = x[(size_t)node * 128 + c] * tanhf(score[node]);
}

__global__ void k_scatter_newid(const int* __restrict__ perm, int* __restrict__ newid, int nNew) {
    int t = blockIdx.x * blockDim.x + threadIdx.x;
    if (t < nNew) newid[perm[t]] = t;
}

__global__ void k_remap_edges(const int* __restrict__ newid, int* __restrict__ src,
                              int* __restrict__ dst, int* __restrict__ em, int E) {
    int t = blockIdx.x * blockDim.x + threadIdx.x;
    if (t >= E) return;
    if (em[t]) {
        int ns = newid[src[t]], nd = newid[dst[t]];
        if (ns >= 0 && nd >= 0) {
            src[t] = ns;
            dst[t] = nd;
        } else {
            src[t] = 0;
            dst[t] = 0;
            em[t]  = 0;
        }
    }
}

// xf[g, 0:128] += max over k rows, xf[g, 128:256] += mean  (block per graph)
__global__ void k_readout(const float* __restrict__ xn, int kk, float* __restrict__ xf) {
    int g = blockIdx.x, c = threadIdx.x;  // blockDim = 128
    float mx = -3.4e38f, sm = 0.f;
    const float* p = xn + (size_t)g * kk * 128 + c;
    for (int r = 0; r < kk; r++) {
        float v = p[(size_t)r * 128];
        mx = v > mx ? v : mx;
        sm += v;
    }
    xf[g * 256 + c] += mx;
    xf[g * 256 + 128 + c] += sm / (float)kk;
}

// tiny MLP head with batchnorm (batch=16), one block
__global__ __launch_bounds__(256) void k_head(const float* __restrict__ xf,
                                              const float* __restrict__ w1, const float* __restrict__ b1,
                                              const float* __restrict__ w2, const float* __restrict__ b2,
                                              const float* __restrict__ w3, const float* __restrict__ b3,
                                              const float* __restrict__ g1, const float* __restrict__ bb1,
                                              const float* __restrict__ g2, const float* __restrict__ bb2,
                                              float* __restrict__ out) {
    __shared__ float sx[16 * 256];
    __shared__ float h1[16 * 128];
    __shared__ float h2[16 * 64];
    int tid = threadIdx.x;
    for (int i = tid; i < 16 * 256; i += 256) sx[i] = xf[i];
    __syncthreads();
    for (int o = tid; o < 16 * 128; o += 256) {
        int r = o >> 7, c = o & 127;
        float s = b1[c];
        for (int kk = 0; kk < 256; kk++) s += sx[r * 256 + kk] * w1[kk * 128 + c];
        h1[o] = s;
    }
    __syncthreads();
    if (tid < 128) {
        int c = tid;
        float m = 0.f;
        for (int r = 0; r < 16; r++) m += h1[r * 128 + c];
        m *= (1.f / 16.f);
        float vv = 0.f;
        for (int r = 0; r < 16; r++) { float d = h1[r * 128 + c] - m; vv += d * d; }
        vv *= (1.f / 16.f);
        float inv = rsqrtf(vv + 1e-5f);
        for (int r = 0; r < 16; r++) {
            float y = (h1[r * 128 + c] - m) * inv * g1[c] + bb1[c];
            h1[r * 128 + c] = y > 0.f ? y : 0.f;
        }
    }
    __syncthreads();
    for (int o = tid; o < 16 * 64; o += 256) {
        int r = o >> 6, c = o & 63;
        float s = b2[c];
        for (int kk = 0; kk < 128; kk++) s += h1[r * 128 + kk] * w2[kk * 64 + c];
        h2[o] = s;
    }
    __syncthreads();
    if (tid < 64) {
        int c = tid;
        float m = 0.f;
        for (int r = 0; r < 16; r++) m += h2[r * 64 + c];
        m *= (1.f / 16.f);
        float vv = 0.f;
        for (int r = 0; r < 16; r++) { float d = h2[r * 64 + c] - m; vv += d * d; }
        vv *= (1.f / 16.f);
        float inv = rsqrtf(vv + 1e-5f);
        for (int r = 0; r < 16; r++) {
            float y = (h2[r * 64 + c] - m) * inv * g2[c] + bb2[c];
            h2[r * 64 + c] = y > 0.f ? y : 0.f;
        }
    }
    __syncthreads();
    if (tid < 16) {
        float s = b3[0];
        for (int kk = 0; kk < 64; kk++) s += h2[tid * 64 + kk] * w3[kk];
        out[tid] = s;
    }
}

// ---------------------------------------------------------------------------
// host launcher
// ---------------------------------------------------------------------------
extern "C" void kernel_launch(void* const* d_in, const int* in_sizes, int n_in,
                              void* d_out, int out_size, void* d_ws, size_t ws_size,
                              hipStream_t stream) {
    const int N0 = 32768, E = 524288, B = 16;
    const int npg[3]  = {2048, 1639, 1312};
    const int ksel[3] = {1639, 1312, 1050};

    const float* x0 = (const float*)d_in[0];
    const long long* ei = (const long long*)d_in[1];
    auto P = [&](int i) { return (const float*)d_in[i]; };
    // param layout (insertion order): 2 + L*18 -> qw qb kw kb vw vb skw skb ow ob
    //                                  ln1g ln1b ln2g ln2b f1w f1b f2w f2b
    // 56 + L*4 -> relw relb rootw rootb ; 68..73 lin1/2/3 w,b ; 74..77 bn g/b

    char* wsp = (char*)d_ws;
    size_t off = 0;
    auto alloc = [&](size_t bytes) {
        off = (off + 255) & ~(size_t)255;
        void* p = wsp + off;
        off += bytes;
        return p;
    };
    const size_t NF = (size_t)N0 * 128 * sizeof(float);
    float* XA   = (float*)alloc(NF);
    float* XB   = (float*)alloc(NF);
    float* Q    = (float*)alloc(NF);
    float* K    = (float*)alloc(NF);
    float* V    = (float*)alloc(NF);
    float* T0   = (float*)alloc(NF);
    float* T1   = (float*)alloc(NF);
    float* HID  = (float*)alloc((size_t)N0 * 64 * sizeof(float));
    float* ALPH = (float*)alloc((size_t)E * 4 * sizeof(float));
    float* DEN  = (float*)alloc((size_t)N0 * 4 * sizeof(float));
    unsigned* AMAX = (unsigned*)alloc((size_t)N0 * 4 * sizeof(unsigned));
    int* SRC   = (int*)alloc((size_t)E * sizeof(int));
    int* DST   = (int*)alloc((size_t)E * sizeof(int));
    int* EM    = (int*)alloc((size_t)E * sizeof(int));
    float* SCR = (float*)alloc((size_t)N0 * sizeof(float));
    int* NEWID = (int*)alloc((size_t)N0 * sizeof(int));
    int* PERM  = (int*)alloc((size_t)N0 * sizeof(int));
    float* XF  = (float*)alloc((size_t)B * 256 * sizeof(float));
    // pre-converted transposed f16 weights: per layer 5x(128*128)+2x(128*64)
    const size_t WT_L = 5 * 16384 + 2 * 8192;  // 98304 halves per layer
    _Float16* WTb = (_Float16*)alloc(3 * WT_L * sizeof(_Float16));
    (void)ws_size; (void)in_sizes; (void)n_in; (void)out_size;

    k_edges_init<<<(E + 255) / 256, 256, 0, stream>>>(ei, SRC, DST, EM, E);
    k_fill_f32<<<(B * 256 + 255) / 256, 256, 0, stream>>>(XF, 0.f, B * 256);

    // one-time weight conversion (f32 [din,dout] -> f16 transposed [dout,din])
    for (int L = 0; L < 3; L++) {
        const int pb = 2 + L * 18;
        _Float16* wt = WTb + L * WT_L;
        k_prep_wt<<<64, 256, 0, stream>>>(P(pb + 0),  wt + 0,     128, 128);  // q
        k_prep_wt<<<64, 256, 0, stream>>>(P(pb + 2),  wt + 16384, 128, 128);  // k
        k_prep_wt<<<64, 256, 0, stream>>>(P(pb + 4),  wt + 32768, 128, 128);  // v
        k_prep_wt<<<64, 256, 0, stream>>>(P(pb + 6),  wt + 49152, 128, 128);  // skip
        k_prep_wt<<<64, 256, 0, stream>>>(P(pb + 8),  wt + 65536, 128, 128);  // out
        k_prep_wt<<<32, 256, 0, stream>>>(P(pb + 14), wt + 81920, 128, 64);   // f1
        k_prep_wt<<<32, 256, 0, stream>>>(P(pb + 16), wt + 90112, 64, 128);   // f2
    }

    const float* xin = x0;
    float* poolOut[3] = {XA, XB, XA};

    for (int L = 0; L < 3; L++) {
        const int n = B * npg[L];
        const int gb128  = (n + 127) / 128;
        const int gbRow  = (n + 7) / 8;
        const int gbEdge = (E + 7) / 8;
        const int gbEH   = (E * 4 + 255) / 256;
        const int pb  = 2 + L * 18;
        const int pbp = 56 + L * 4;
        _Float16* wt = WTb + L * WT_L;

        // q, k, v, skip projections (WMMA, TDM-staged weights)
        k_linear_wmma<128, 128, false><<<gb128, 256, 0, stream>>>(xin, wt + 0,     P(pb + 1), Q,  n);
        k_linear_wmma<128, 128, false><<<gb128, 256, 0, stream>>>(xin, wt + 16384, P(pb + 3), K,  n);
        k_linear_wmma<128, 128, false><<<gb128, 256, 0, stream>>>(xin, wt + 32768, P(pb + 5), V,  n);
        k_linear_wmma<128, 128, false><<<gb128, 256, 0, stream>>>(xin, wt + 49152, P(pb + 7), T0, n);

        // edge softmax attention, accumulated on top of skip (T0)
        k_fill_u32<<<(n * 4 + 255) / 256, 256, 0, stream>>>(AMAX, 0x007FFFFFu, n * 4);  // enc(-inf)
        k_fill_f32<<<(n * 4 + 255) / 256, 256, 0, stream>>>(DEN, 0.f, n * 4);
        k_attn_alpha<<<gbEH, 256, 0, stream>>>(Q, K, SRC, DST, EM, ALPH, AMAX, E);
        k_attn_exp<<<gbEH, 256, 0, stream>>>(ALPH, AMAX, DEN, DST, EM, E);
        k_attn_scatter<<<gbEdge, 256, 0, stream>>>(ALPH, DEN, V, SRC, DST, EM, T0, E);

        // out projection, residual LN, FFN, residual LN
        k_linear_wmma<128, 128, false><<<gb128, 256, 0, stream>>>(T0, wt + 65536, P(pb + 9), T1, n);
        k_ln_res<<<gbRow, 256, 0, stream>>>(T1, xin, P(pb + 10), P(pb + 11), Q, n);  // o2 -> Q
        k_linear_wmma<128, 64, true><<<gb128, 256, 0, stream>>>(Q, wt + 81920, P(pb + 15), HID, n);
        k_linear_wmma<64, 128, false><<<gb128, 256, 0, stream>>>(HID, wt + 90112, P(pb + 17), T1, n);
        k_ln_res<<<gbRow, 256, 0, stream>>>(T1, Q, P(pb + 12), P(pb + 13), V, n);  // gt out -> V

        // SAGPool
        k_fill_f32<<<(int)(((size_t)n * 128 + 255) / 256), 256, 0, stream>>>(K, 0.f, n * 128);
        k_pool_scatter<<<gbEdge, 256, 0, stream>>>(V, SRC, DST, EM, K, E);
        k_score<<<gbRow, 256, 0, stream>>>(K, V, P(pbp + 0), P(pbp + 1), P(pbp + 2), P(pbp + 3), SCR, n);
        k_topk<<<B, 256, 0, stream>>>(SCR, npg[L], ksel[L], PERM);
        const int nNew = B * ksel[L];
        k_gather_tanh<<<(nNew * 128 + 255) / 256, 256, 0, stream>>>(V, SCR, PERM, poolOut[L], nNew);
        k_fill_i32<<<(n + 255) / 256, 256, 0, stream>>>(NEWID, -1, n);
        k_scatter_newid<<<(nNew + 255) / 256, 256, 0, stream>>>(PERM, NEWID, nNew);
        k_remap_edges<<<(E + 255) / 256, 256, 0, stream>>>(NEWID, SRC, DST, EM, E);
        k_readout<<<B, 128, 0, stream>>>(poolOut[L], ksel[L], XF);

        xin = poolOut[L];
    }

    k_head<<<1, 256, 0, stream>>>(XF, P(68), P(69), P(70), P(71), P(72), P(73),
                                  P(74), P(75), P(76), P(77), (float*)d_out);
}
